// GlobalContextAttentionDual_58720792871194
// MI455X (gfx1250) — matrix operand order repaired and checked by reading, not compiled, not run on hardware
//
#include <hip/hip_runtime.h>

typedef __bf16 bf16_t;
typedef __attribute__((ext_vector_type(16))) __bf16 v16bf;
typedef __attribute__((ext_vector_type(8)))  __bf16 v8bf;
typedef __attribute__((ext_vector_type(2)))  __bf16 v2bf;
typedef __attribute__((ext_vector_type(8)))  float  v8f;

static __device__ __forceinline__ v16bf cat8(v8bf lo, v8bf hi) {
  v16bf r;
#pragma unroll
  for (int i = 0; i < 8; ++i) { r[i] = lo[i]; r[i + 8] = hi[i]; }
  return r;
}
static __device__ __forceinline__ v8f vzero8() {
  v8f r;
#pragma unroll
  for (int i = 0; i < 8; ++i) r[i] = 0.0f;
  return r;
}

#define WMMA_BF16(a, b, c) \
  __builtin_amdgcn_wmma_f32_16x16x32_bf16(false, (a), false, (b), (short)0, (c), false, false)

// CDNA5 async DMA: global -> LDS, 16B per lane, tracked by ASYNCcnt.
static __device__ __forceinline__ void async_load_b128(unsigned lds_off, const bf16_t* g) {
  asm volatile("global_load_async_to_lds_b128 %0, %1, off"
               :: "v"(lds_off), "v"((unsigned long long)(size_t)g)
               : "memory");
}
static __device__ __forceinline__ void wait_async0() {
  asm volatile("s_wait_asynccnt 0x0" ::: "memory");
}

// 16-lane butterfly reductions in pure VALU via DPP16 (no ds_bpermute, no
// dscnt stalls): xor1 = quad_perm[1,0,3,2], xor2 = quad_perm[2,3,0,1],
// then row_half_mirror (pairs quads within 8) and row_mirror (pairs 8-groups).
template <int CTRL>
static __device__ __forceinline__ float dppmove(float x) {
  return __int_as_float(__builtin_amdgcn_update_dpp(
      0, __float_as_int(x), CTRL, 0xF, 0xF, true));
}
static __device__ __forceinline__ float red16_max(float x) {
  x = fmaxf(x, dppmove<0xB1>(x));
  x = fmaxf(x, dppmove<0x4E>(x));
  x = fmaxf(x, dppmove<0x141>(x));  // ROW_HALF_MIRROR
  x = fmaxf(x, dppmove<0x140>(x));  // ROW_MIRROR
  return x;
}
static __device__ __forceinline__ float red16_sum(float x) {
  x += dppmove<0xB1>(x);
  x += dppmove<0x4E>(x);
  x += dppmove<0x141>(x);
  x += dppmove<0x140>(x);
  return x;
}

// ---------------------------------------------------------------------------
// out = (A @ W^T) * outScale (+ bias). A: [M][K] fp32/bf16, W: [N][K] fp32.
// Block = 128 threads (4 waves), tile 64(M) x 128(N), K-step 32.
// Each wave computes 32x64 = 2x4 WMMA tiles -> 8 v_wmma per K-step.
// ---------------------------------------------------------------------------
template <typename InT, bool ADD_BIAS, typename OutT>
__global__ __launch_bounds__(128) void gemm_bt_wmma(
    const InT* __restrict__ A, const float* __restrict__ W,
    const float* __restrict__ bias, OutT* __restrict__ out,
    float outScale, int M, int N, int K)
{
  __shared__ __align__(16) bf16_t Asub[64][40];   // [m][k]
  __shared__ __align__(16) bf16_t Bsub[128][40];  // [n][k]

  const int tid  = threadIdx.x;
  const int lane = tid & 31;
  const int wv   = tid >> 5;
  const int m0 = blockIdx.x * 64;
  const int n0 = blockIdx.y * 128;
  const int wm = (wv & 1) * 32;
  const int wn = (wv >> 1) * 64;
  const int hf = lane >> 4;
  const int mr = lane & 15;

  v8f acc[2][4];
#pragma unroll
  for (int a = 0; a < 2; ++a)
#pragma unroll
    for (int b = 0; b < 4; ++b) acc[a][b] = vzero8();

  for (int k0 = 0; k0 < K; k0 += 32) {
    {  // stage A tile: 64x32, 16 elems/thread -> 2 ds_store_b128
      const int r = tid >> 1, c = (tid & 1) * 16;
      const InT* aRow = A + (size_t)(m0 + r) * K + k0 + c;
      v8bf t0, t1;
#pragma unroll
      for (int i = 0; i < 8; ++i) { t0[i] = (bf16_t)(float)aRow[i]; t1[i] = (bf16_t)(float)aRow[8 + i]; }
      *(v8bf*)&Asub[r][c] = t0;
      *(v8bf*)&Asub[r][c + 8] = t1;
      if (k0 + 32 < K) __builtin_prefetch(aRow + 32, 0, 0);
    }
    {  // stage B tile: 128x32, 32 elems/thread -> 4 ds_store_b128
      const float* wRow = W + (size_t)(n0 + tid) * K + k0;
#pragma unroll
      for (int j = 0; j < 4; ++j) {
        v8bf t;
#pragma unroll
        for (int i = 0; i < 8; ++i) t[i] = (bf16_t)wRow[j * 8 + i];
        *(v8bf*)&Bsub[tid][j * 8] = t;
      }
      if (k0 + 32 < K) __builtin_prefetch(wRow + 32, 0, 0);
    }
    __syncthreads();

    v16bf aF[2], bF[4];
#pragma unroll
    for (int tm = 0; tm < 2; ++tm) {
      const bf16_t* p = &Asub[wm + tm * 16 + mr][hf * 8];
      aF[tm] = cat8(*(const v8bf*)p, *(const v8bf*)(p + 16));
    }
#pragma unroll
    for (int tn = 0; tn < 4; ++tn) {
      const bf16_t* p = &Bsub[wn + tn * 16 + mr][hf * 8];
      bF[tn] = cat8(*(const v8bf*)p, *(const v8bf*)(p + 16));
    }
#pragma unroll
    for (int tm = 0; tm < 2; ++tm)
#pragma unroll
      for (int tn = 0; tn < 4; ++tn)
        acc[tm][tn] = WMMA_BF16(aF[tm], bF[tn], acc[tm][tn]);
    __syncthreads();
  }

  // C/D layout: VGPR i, lanes 0-15 -> M=i, lanes 16-31 -> M=8+i; N = lane&15
#pragma unroll
  for (int tm = 0; tm < 2; ++tm)
#pragma unroll
    for (int tn = 0; tn < 4; ++tn) {
      const int col = n0 + wn + tn * 16 + mr;
      const float bv = ADD_BIAS ? bias[col] : 0.0f;
#pragma unroll
      for (int i = 0; i < 8; ++i) {
        const int row = m0 + wm + tm * 16 + hf * 8 + i;
        out[(size_t)row * N + col] = (OutT)(acc[tm][tn][i] * outScale + bv);
      }
    }
}

// ---------------------------------------------------------------------------
// Vw[b,c,:] = V[b,c,:] * wS[b,c]  (bf16)
// ---------------------------------------------------------------------------
__global__ __launch_bounds__(256) void scale_v_kernel(
    const bf16_t* __restrict__ Vp, const float* __restrict__ wS,
    bf16_t* __restrict__ Vw)
{
  const size_t i = (size_t)blockIdx.x * blockDim.x + threadIdx.x;
  Vw[i] = (bf16_t)((float)Vp[i] * wS[i >> 10]);  // row = i / D, D = 1024
}

// ---------------------------------------------------------------------------
// Fused flash-style dual attention.
// Grid: (Nt/128, B*H). Block 256 = 8 waves; wave w owns query rows [w*16,+16).
// 128-row query tile halves K/V HBM traffic vs 64-row tiles.
// Q/K tiles staged via global_load_async_to_lds_b128 (ASYNCcnt DMA).
// K-dim of the P*V step uses interleaved column order j = 2n+cs (order-
// invariant under summation), letting P and V-pair stores be dword-packed.
// ---------------------------------------------------------------------------
__global__ __launch_bounds__(256) void attn_dual_kernel(
    const bf16_t* __restrict__ Qp, const bf16_t* __restrict__ Kp,
    const bf16_t* __restrict__ Vp, const bf16_t* __restrict__ Vw,
    const unsigned char* __restrict__ mask,
    bf16_t* __restrict__ Cpos, bf16_t* __restrict__ Cneg)
{
  constexpr int NT = 2048, NC = 2048, D = 1024, DK = 64;
  __shared__ __align__(16) bf16_t Qs[128][72];    // [q][d] stride 144B
  __shared__ __align__(16) bf16_t Ks[32][72];     // [c][d]
  __shared__ __align__(16) bf16_t Vt[64][40];     // [d][j]  j = 2n+cs (c = n+16*cs)
  __shared__ __align__(16) bf16_t Vwt[64][40];    // [d][j]
  __shared__ __align__(16) bf16_t Pw[8][16][40];  // per-wave P scratch [m][j]

  const int tid  = threadIdx.x;
  const int lane = tid & 31;
  const int wv   = tid >> 5;
  const int hf   = lane >> 4;
  const int mr   = lane & 15;
  const int q0 = blockIdx.x * 128;
  const int b  = blockIdx.y >> 4;
  const int h  = blockIdx.y & 15;

  // --- stage Q tile (pre-scaled by 1/8 in the projection GEMM) via async DMA
  {
    const unsigned qbase = (unsigned)(size_t)&Qs[0][0];
    const bf16_t* gq = Qp + ((size_t)b * NT + q0) * D + h * DK;
#pragma unroll
    for (int j = 0; j < 4; ++j) {
      const int s = tid * 4 + j;              // 0..1023 16B segments
      const int r = s >> 3, c8 = s & 7;
      async_load_b128(qbase + (unsigned)(r * 144 + c8 * 16), gq + (size_t)r * D + c8 * 8);
    }
    wait_async0();
  }
  __syncthreads();

  v16bf aQ[2];
#pragma unroll
  for (int ks = 0; ks < 2; ++ks) {
    const bf16_t* p = &Qs[wv * 16 + mr][ks * 32 + hf * 8];
    aQ[ks] = cat8(*(const v8bf*)p, *(const v8bf*)(p + 16));
  }

  v8f acc_all[4], acc_pos[4];
#pragma unroll
  for (int t = 0; t < 4; ++t) { acc_all[t] = vzero8(); acc_pos[t] = vzero8(); }
  float rmax[8], rsum[8];
#pragma unroll
  for (int i = 0; i < 8; ++i) { rmax[i] = -1e30f; rsum[i] = 0.0f; }

  const unsigned kbase = (unsigned)(size_t)&Ks[0][0];
  const int vn  = tid & 15;          // column pair (c = vn, vn+16)
  const int vd0 = (tid >> 4) * 8;    // 8 d-values (threads 0..127 only)

  for (int c0 = 0; c0 < NC; c0 += 32) {
    __syncthreads();
    {  // stage K chunk [32][64] via async DMA: 1 x 16B segment / thread
      const bf16_t* gk = Kp + ((size_t)b * NC + c0) * D + h * DK;
      const int r = tid >> 3, c8 = tid & 7;   // 256 segments
      async_load_b128(kbase + (unsigned)(r * 144 + c8 * 16), gk + (size_t)r * D + c8 * 8);
    }
    if (tid < 128) {  // stage V / V*wS transposed+interleaved, dword-packed
      const bf16_t* g0 = Vp + ((size_t)b * NC + c0 + vn) * D + h * DK + vd0;
      const bf16_t* g1 = g0 + (size_t)16 * D;
      const bf16_t* w0 = Vw + ((size_t)b * NC + c0 + vn) * D + h * DK + vd0;
      const bf16_t* w1 = w0 + (size_t)16 * D;
      const v8bf a0 = *(const v8bf*)g0, a1 = *(const v8bf*)g1;
      const v8bf b0 = *(const v8bf*)w0, b1 = *(const v8bf*)w1;
#pragma unroll
      for (int i = 0; i < 8; ++i) {
        v2bf tv; tv[0] = a0[i]; tv[1] = a1[i];
        v2bf tw; tw[0] = b0[i]; tw[1] = b1[i];
        *(v2bf*)&Vt[vd0 + i][vn * 2]  = tv;
        *(v2bf*)&Vwt[vd0 + i][vn * 2] = tw;
      }
    }
    wait_async0();
    __syncthreads();

    // S = Q(16x64) x K^T -> two 16x16 tiles (columns c = c0 + cs*16 + n)
    float s[2][8];
#pragma unroll
    for (int cs = 0; cs < 2; ++cs) {
      v8f sv = vzero8();
#pragma unroll
      for (int ks = 0; ks < 2; ++ks) {
        const bf16_t* p = &Ks[cs * 16 + mr][ks * 32 + hf * 8];
        v16bf bK = cat8(*(const v8bf*)p, *(const v8bf*)(p + 16));
        sv = WMMA_BF16(aQ[ks], bK, sv);
      }
      const unsigned char mk = mask[(size_t)b * NC + c0 + cs * 16 + mr];
      const float mbias = mk ? 0.0f : -1e30f;
#pragma unroll
      for (int i = 0; i < 8; ++i) s[cs][i] = sv[i] + mbias;
    }

    // Online softmax: rows in VGPR index i, columns across the 16-lane group.
    // Reductions are DPP16 butterflies (pure VALU).
    float pr0[8], pr1[8], scl[8];
#pragma unroll
    for (int i = 0; i < 8; ++i) {
      const float rm = red16_max(fmaxf(s[0][i], s[1][i]));
      const float mn = fmaxf(fmaxf(rmax[i], rm), -1e29f);
      scl[i] = __expf(rmax[i] - mn);
      pr0[i] = __expf(s[0][i] - mn);
      pr1[i] = __expf(s[1][i] - mn);
      const float ps = red16_sum(pr0[i] + pr1[i]);
      rsum[i] = rsum[i] * scl[i] + ps;
      rmax[i] = mn;
    }
#pragma unroll
    for (int t = 0; t < 4; ++t)
#pragma unroll
      for (int i = 0; i < 8; ++i) { acc_all[t][i] *= scl[i]; acc_pos[t][i] *= scl[i]; }

    // P (C-layout) -> LDS (interleaved j = 2n+cs, packed b32) -> A-fragment
#pragma unroll
    for (int i = 0; i < 8; ++i) {
      v2bf tp; tp[0] = (bf16_t)pr0[i]; tp[1] = (bf16_t)pr1[i];
      *(v2bf*)&Pw[wv][hf * 8 + i][mr * 2] = tp;
    }
    const bf16_t* pp = &Pw[wv][mr][hf * 8];
    v16bf aP = cat8(*(const v8bf*)pp, *(const v8bf*)(pp + 16));

    // acc += P(16x32) x V(32x64) for both V and V*wS (K order matches j)
#pragma unroll
    for (int t = 0; t < 4; ++t) {
      const bf16_t* pv = &Vt[t * 16 + mr][hf * 8];
      const bf16_t* pw = &Vwt[t * 16 + mr][hf * 8];
      v16bf bV = cat8(*(const v8bf*)pv, *(const v8bf*)(pv + 16));
      v16bf bW = cat8(*(const v8bf*)pw, *(const v8bf*)(pw + 16));
      acc_all[t] = WMMA_BF16(aP, bV, acc_all[t]);
      acc_pos[t] = WMMA_BF16(aP, bW, acc_pos[t]);
    }
  }

  // Finalize: ctx_pos = acc_pos/l ; ctx_neg = (acc_all - acc_pos)/l
#pragma unroll
  for (int i = 0; i < 8; ++i) {
    const float inv = rsum[i] > 0.0f ? 1.0f / rsum[i] : 0.0f;
    const int row = q0 + wv * 16 + hf * 8 + i;
#pragma unroll
    for (int t = 0; t < 4; ++t) {
      const int col = h * DK + t * 16 + mr;
      const float all = acc_all[t][i] * inv;
      const float pos = acc_pos[t][i] * inv;
      const size_t idx = ((size_t)b * NT + row) * D + col;
      Cpos[idx] = (bf16_t)pos;
      Cneg[idx] = (bf16_t)(all - pos);
    }
  }
}

// ---------------------------------------------------------------------------
extern "C" void kernel_launch(void* const* d_in, const int* in_sizes, int n_in,
                              void* d_out, int out_size, void* d_ws, size_t ws_size,
                              hipStream_t stream) {
  (void)in_sizes; (void)n_in; (void)out_size; (void)ws_size;
  const float* Qsrc = (const float*)d_in[0];
  const float* Ksrc = (const float*)d_in[1];
  const float* Vsrc = (const float*)d_in[2];
  const float* wS   = (const float*)d_in[3];
  const unsigned char* mask = (const unsigned char*)d_in[4];
  const float* Wq = (const float*)d_in[5];
  const float* Wk = (const float*)d_in[6];
  const float* Wv = (const float*)d_in[7];
  const float* Wo = (const float*)d_in[8];
  const float* bo = (const float*)d_in[9];

  const size_t E = (size_t)4096 * 1024;  // B*N*D elements
  bf16_t* Qp = (bf16_t*)d_ws;            // 6 bf16 buffers x 8MB = 48MB
  bf16_t* Kp = Qp + E;
  bf16_t* Vp = Kp + E;
  bf16_t* Vw = Vp + E;
  bf16_t* Cp = Vw + E;
  bf16_t* Cn = Cp + E;

  const dim3 gG(64, 8), blk(128);
  // Q is pre-scaled by 1/sqrt(d_k) = 0.125 here so attention staging is pure DMA.
  gemm_bt_wmma<float,  false, bf16_t><<<gG, blk, 0, stream>>>(Qsrc, Wq, nullptr, Qp, 0.125f, 4096, 1024, 1024);
  gemm_bt_wmma<float,  false, bf16_t><<<gG, blk, 0, stream>>>(Ksrc, Wk, nullptr, Kp, 1.0f,   4096, 1024, 1024);
  gemm_bt_wmma<float,  false, bf16_t><<<gG, blk, 0, stream>>>(Vsrc, Wv, nullptr, Vp, 1.0f,   4096, 1024, 1024);

  scale_v_kernel<<<dim3((unsigned)(E / 256)), 256, 0, stream>>>(Vp, wS, Vw);

  attn_dual_kernel<<<dim3(16, 32), 256, 0, stream>>>(Qp, Kp, Vp, Vw, mask, Cp, Cn);

  float* out = (float*)d_out;
  gemm_bt_wmma<bf16_t, true,  float ><<<gG, blk, 0, stream>>>(Cp, Wo, bo, out,     1.0f, 4096, 1024, 1024);
  gemm_bt_wmma<bf16_t, true,  float ><<<gG, blk, 0, stream>>>(Cn, Wo, bo, out + E, 1.0f, 4096, 1024, 1024);
}
